// NGCF_60318520705223
// MI455X (gfx1250) — matrix operand order repaired and checked by reading, not compiled, not run on hardware
//
#include <hip/hip_runtime.h>
#include <hip/hip_bf16.h>

#define N_NODES 100000
#define N_EDGES 1600000
#define DIM     128
#define KLAYERS 3

typedef __attribute__((ext_vector_type(16))) __bf16       v16bf;
typedef __attribute__((ext_vector_type(8)))  float        v8f;
typedef __attribute__((ext_vector_type(8)))  unsigned int v8u;
typedef __attribute__((ext_vector_type(4)))  unsigned int v4u;
typedef __attribute__((ext_vector_type(8)))  int          v8i;
typedef __attribute__((ext_vector_type(4)))  int          v4i;

static __device__ __forceinline__ v16bf bc16(v8u u) {
    return __builtin_bit_cast(v16bf, u);
}

static __device__ __forceinline__ unsigned int pack2bf(float a, float b) {
    __bf16 b0 = (__bf16)a;
    __bf16 b1 = (__bf16)b;
    unsigned int lo = (unsigned int)__builtin_bit_cast(unsigned short, b0);
    unsigned int hi = (unsigned int)__builtin_bit_cast(unsigned short, b1);
    return lo | (hi << 16);
}

// ---------------------------------------------------------------------------
// Repack W_gc / W_bi (f32, [3][128][128], row = K, col = N) into WMMA B-operand
// layout: per (matrix m, kstep kk, ntile t, lane) -> 8 dwords (16 bf16).
// Lanes 0-15 hold K = kk*32 + 0..15 (pairs per dword), lanes 16-31 K = +16.
// ---------------------------------------------------------------------------
__global__ void pack_weights(const float* __restrict__ Wgc,
                             const float* __restrict__ Wbi,
                             unsigned int* __restrict__ Bpack) {
    int tid = blockIdx.x * blockDim.x + threadIdx.x;
    if (tid >= 6 * 1024) return;
    int m    = tid >> 10;        // 0..5 : layer*2 + (0=gc,1=bi)
    int r    = tid & 1023;
    int kk   = r >> 8;           // 0..3
    int t    = (r >> 5) & 7;     // 0..7
    int lane = r & 31;
    const float* W = (m & 1) ? (Wbi + (size_t)(m >> 1) * DIM * DIM)
                             : (Wgc + (size_t)(m >> 1) * DIM * DIM);
    int n     = t * 16 + (lane & 15);
    int kbase = ((lane < 16) ? 0 : 16) + kk * 32;
    unsigned int* dst = Bpack + (size_t)(((m * 4 + kk) * 8 + t) * 32 + lane) * 8;
#pragma unroll
    for (int v = 0; v < 8; ++v) {
        int k0 = kbase + 2 * v;
        dst[v] = pack2bf(W[(size_t)k0 * DIM + n], W[(size_t)(k0 + 1) * DIM + n]);
    }
}

__global__ void bias_sum(const float* __restrict__ bgc,
                         const float* __restrict__ bbi,
                         float* __restrict__ bsum) {
    int tid = blockIdx.x * blockDim.x + threadIdx.x;
    if (tid < KLAYERS * DIM) bsum[tid] = bgc[tid] + bbi[tid];
}

// layer-0 slice of the output: out[:, 0:128] = embeddings
__global__ void copy_layer0(const float* __restrict__ emb, float* __restrict__ out) {
    int tid = blockIdx.x * blockDim.x + threadIdx.x;   // N*32 float4 moves
    int r = tid >> 5;
    int q = tid & 31;
    if (r < N_NODES) {
        const float4* s = (const float4*)(emb + (size_t)r * DIM);
        float4* d = (float4*)(out + (size_t)r * (4 * DIM));
        d[q] = s[q];
    }
}

__global__ void zero_f32(float4* __restrict__ p, int n4) {
    int tid = blockIdx.x * blockDim.x + threadIdx.x;
    if (tid < n4) p[tid] = make_float4(0.f, 0.f, 0.f, 0.f);
}

// ---------------------------------------------------------------------------
// SPMM: one wave per edge; lane owns 4 consecutive features (float4).
// f32 atomic scatter-add into Lx (L2-resident: x and Lx both fit in 192MB L2).
// ---------------------------------------------------------------------------
__global__ void __launch_bounds__(256)
spmm(const int* __restrict__ row, const int* __restrict__ col,
     const float* __restrict__ val, const float* __restrict__ x,
     float* __restrict__ Lx) {
    int lane = threadIdx.x & 31;
    int e = blockIdx.x * 8 + (threadIdx.x >> 5);
    if (e >= N_EDGES) return;
    int r = row[e];
    int c = col[e];
    float w = val[e];
    float4 xv = ((const float4*)(x + (size_t)c * DIM))[lane];
    float* dst = Lx + (size_t)r * DIM + lane * 4;
    __hip_atomic_fetch_add(dst + 0, w * xv.x, __ATOMIC_RELAXED, __HIP_MEMORY_SCOPE_AGENT);
    __hip_atomic_fetch_add(dst + 1, w * xv.y, __ATOMIC_RELAXED, __HIP_MEMORY_SCOPE_AGENT);
    __hip_atomic_fetch_add(dst + 2, w * xv.z, __ATOMIC_RELAXED, __HIP_MEMORY_SCOPE_AGENT);
    __hip_atomic_fetch_add(dst + 3, w * xv.w, __ATOMIC_RELAXED, __HIP_MEMORY_SCOPE_AGENT);
}

// ---------------------------------------------------------------------------
// Fused GEMM: D = (Lx+x)@Wgc + (x*Lx)@Wbi + bias -> leakyReLU -> L2-normalize.
// A-operands are formed in-register from f32 x/Lx (prep pass eliminated).
// Both packed B matrices (64KB) are staged to LDS by the Tensor Data Mover.
// One wave computes a 16x128 strip: 4 ksteps x 8 ntiles x 2 matrices = 64 WMMAs.
// ---------------------------------------------------------------------------
__global__ void __launch_bounds__(256)
gemm_layer(const float* __restrict__ x, const float* __restrict__ Lx,
           const unsigned int* __restrict__ Bpk,   // layer base: gc(32KB)+bi(32KB)
           const float* __restrict__ bsum, float* __restrict__ outp,
           float* __restrict__ xnext) {
    __shared__ __align__(16) unsigned char shB[65536];

    // ---- stage 64KB of packed weights into LDS ----
#if __has_builtin(__builtin_amdgcn_tensor_load_to_lds)
    if (threadIdx.x < 32) {                       // one wave issues the TDM
        unsigned lds_base = (unsigned)(size_t)(void*)shB;   // low 32 bits = LDS offset
        unsigned long long ga = (unsigned long long)(size_t)Bpk;
        v4u g0;
        g0[0] = 1u;                                          // count=1, gather off
        g0[1] = lds_base;                                    // lds_addr
        g0[2] = (unsigned)(ga & 0xffffffffu);                // global_addr[31:0]
        g0[3] = (unsigned)((ga >> 32) & 0x01ffffffu) | 0x80000000u; // [56:32]|type=2
        v8i g1;
        g1[0] = 0x00030000;          // wg_mask=0, data_size=3 (8B elements)
        g1[1] = (int)(0x2000u << 16);// tensor_dim0 = 8192 elems (64KB), low 16 bits
        g1[2] = (int)(1u << 16);     // tensor_dim0 hi=0 | tensor_dim1 = 1 (lo)
        g1[3] = (int)(0x2000u << 16);// tensor_dim1 hi=0 | tile_dim0 = 8192
        g1[4] = 1;                   // tile_dim1 = 1, tile_dim2 = 0
        g1[5] = 0x2000;              // tensor_dim0_stride = 8192 (lo 32)
        g1[6] = 0;                   // stride hi, tensor_dim1_stride lo
        g1[7] = 0;
        v4i gz = {0, 0, 0, 0};
#if defined(__clang_major__) && (__clang_major__ >= 23)
        v8i gz8 = {0, 0, 0, 0, 0, 0, 0, 0};
        __builtin_amdgcn_tensor_load_to_lds(g0, g1, gz, gz, gz8, 0);
#else
        __builtin_amdgcn_tensor_load_to_lds(g0, g1, gz, gz, 0);
#endif
        __builtin_amdgcn_s_wait_tensorcnt(0);
    }
    __syncthreads();
#else
    {   // fallback: cooperative staging
        const float4* src = (const float4*)Bpk;
        float4* dst = (float4*)shB;
#pragma unroll
        for (int i = 0; i < 16; ++i)
            dst[threadIdx.x + i * 256] = src[threadIdx.x + i * 256];
        __syncthreads();
    }
#endif

    const int NT = N_NODES / 16;                  // 6250, exact
    int lane = threadIdx.x & 31;
    int mtile = blockIdx.x * 8 + (threadIdx.x >> 5);
    if (mtile >= NT) mtile = NT - 1;              // clamp: redundant identical work
    int rlo = lane & 15;

    size_t rowoff = (size_t)(mtile * 16 + rlo) * DIM;
    const float* xr = x + rowoff;
    const float* lr = Lx + rowoff;
    int hi8 = (lane >= 16) ? 16 : 0;              // upper half-wave holds K+16
    const v8u* bls = (const v8u*)shB;             // gc: [0,1024), bi: [1024,2048)

    v8f acc[8] = {};
#pragma unroll
    for (int kk = 0; kk < 4; ++kk) {
        v8u ua, uh;
#pragma unroll
        for (int v = 0; v < 8; ++v) {
            // A layout: vgpr v -> K pair base 2v; +16 upper half
            int k0 = kk * 32 + 2 * v + hi8;
            float2 xv = *(const float2*)(xr + k0);
            float2 lv = *(const float2*)(lr + k0);
            ua[v] = pack2bf(lv.x + xv.x, lv.y + xv.y);   // prop = Lx + x
            uh[v] = pack2bf(xv.x * lv.x, xv.y * lv.y);   // h    = x * Lx
        }
        v16bf aP = bc16(ua);
        v16bf aH = bc16(uh);
#pragma unroll
        for (int t = 0; t < 8; ++t) {
            v16bf bgv = bc16(bls[(kk * 8 + t) * 32 + lane]);
            v16bf bbv = bc16(bls[1024 + (kk * 8 + t) * 32 + lane]);
            acc[t] = __builtin_amdgcn_wmma_f32_16x16x32_bf16(
                false, aP, false, bgv, (short)0, acc[t], false, false);
            acc[t] = __builtin_amdgcn_wmma_f32_16x16x32_bf16(
                false, aH, false, bbv, (short)0, acc[t], false, false);
        }
    }

    // Epilogue: bias + leaky relu + per-row sum of squares.
    // C layout: vgpr j, lanes 0-15 -> row j, col=lane; lanes 16-31 -> row j+8.
    float ssq[8];
#pragma unroll
    for (int j = 0; j < 8; ++j) ssq[j] = 0.f;
#pragma unroll
    for (int t = 0; t < 8; ++t) {
        float bias = bsum[t * 16 + rlo];
#pragma unroll
        for (int j = 0; j < 8; ++j) {
            float v = acc[t][j] + bias;
            v = (v > 0.f) ? v : 0.2f * v;
            acc[t][j] = v;
            ssq[j] += v * v;
        }
    }
#pragma unroll
    for (int j = 0; j < 8; ++j) {
        float s = ssq[j];
        s += __shfl_xor(s, 1, 16);     // reduce within each 16-lane half
        s += __shfl_xor(s, 2, 16);
        s += __shfl_xor(s, 4, 16);
        s += __shfl_xor(s, 8, 16);
        ssq[j] = rsqrtf(fmaxf(s, 1e-12f));
    }
    int rbase = mtile * 16 + ((lane >= 16) ? 8 : 0);
#pragma unroll
    for (int j = 0; j < 8; ++j) {
        size_t r = (size_t)(rbase + j);
        float s = ssq[j];
#pragma unroll
        for (int t = 0; t < 8; ++t) {
            float v = acc[t][j] * s;
            int c = t * 16 + rlo;
            xnext[r * DIM + c] = v;
            outp[r * (4 * DIM) + c] = v;    // output slice for this depth
        }
    }
}

extern "C" void kernel_launch(void* const* d_in, const int* in_sizes, int n_in,
                              void* d_out, int out_size, void* d_ws, size_t ws_size,
                              hipStream_t stream) {
    const float* emb  = (const float*)d_in[0];
    const int*   erow = (const int*)d_in[1];
    const int*   ecol = (const int*)d_in[2];
    const float* eval = (const float*)d_in[3];
    const float* Wgc  = (const float*)d_in[4];
    const float* bgc  = (const float*)d_in[5];
    const float* Wbi  = (const float*)d_in[6];
    const float* bbi  = (const float*)d_in[7];
    float* out = (float*)d_out;

    char* ws = (char*)d_ws;
    size_t o = 0;
    float* Lx   = (float*)(ws + o);        o += (size_t)N_NODES * DIM * 4;  // 51.2MB
    float* xbuf = (float*)(ws + o);        o += (size_t)N_NODES * DIM * 4;  // 51.2MB
    unsigned int* Bpack = (unsigned int*)(ws + o); o += (size_t)6 * 1024 * 8 * 4;
    float* bsum = (float*)(ws + o);        o += (size_t)KLAYERS * DIM * 4;

    pack_weights<<<(6 * 1024 + 255) / 256, 256, 0, stream>>>(Wgc, Wbi, Bpack);
    bias_sum<<<2, 256, 0, stream>>>(bgc, bbi, bsum);
    copy_layer0<<<(N_NODES * 32 + 255) / 256, 256, 0, stream>>>(emb, out);

    for (int i = 0; i < KLAYERS; ++i) {
        const float* xcur = (i == 0) ? emb : xbuf;
        zero_f32<<<(N_NODES * DIM / 4 + 255) / 256, 256, 0, stream>>>((float4*)Lx,
                                                                      N_NODES * DIM / 4);
        spmm<<<(N_EDGES + 7) / 8, 256, 0, stream>>>(erow, ecol, eval, xcur, Lx);
        // layer's packed weights: gc then bi, contiguous 64KB
        const unsigned int* Bpk = Bpack + (size_t)(2 * i) * 8192;
        gemm_layer<<<(N_NODES / 16 + 7) / 8, 256, 0, stream>>>(
            xcur, Lx, Bpk, bsum + (size_t)i * DIM,
            out + (size_t)(i + 1) * DIM, xbuf);
    }
}
